// CausalSelfAttention_62225486184511
// MI455X (gfx1250) — matrix operand order repaired and checked by reading, not compiled, not run on hardware
//
#include <hip/hip_runtime.h>
#include <hip/hip_bf16.h>

// ---------------------------------------------------------------------------
// Causal self-attention for MI455X (gfx1250), wave32 + WMMA bf16.
//   q = x Wq^T ; k = x Wk^T ; v = x Wv^T   (bf16 WMMA, f32 accum)
//   flash-attention with online softmax    (WMMA for QK^T and PV)
//   out = attn Wo^T                        (bf16 WMMA, f32 out)
// GEMMs: 32x64 wave tiles, k-loop unrolled x2 with alternating fragment
// buffers (copy-free software pipeline: loads land directly in the registers
// consumed one stage later).
// ---------------------------------------------------------------------------

typedef __bf16 bf16_t;
typedef __attribute__((ext_vector_type(8)))  __bf16 v8bf;
typedef __attribute__((ext_vector_type(16))) __bf16 v16bf;
typedef __attribute__((ext_vector_type(8)))  float  v8f;

#define S_LEN  2048
#define BATCH  4
#define DMODEL 768
#define NHEADS 12
#define HDIM   64
#define NTOK   (S_LEN * BATCH)      // 8192 rows in the token dimension

__device__ __forceinline__ v8bf load8(const bf16_t* p) {
    return *reinterpret_cast<const v8bf*>(p);
}

// A-fragment (16x32, MxK) of a row-major bf16 matrix, tile origin (row0,k0).
// CDNA5 layout: lane m=lane&15 holds row m; lanes 0-15 hold K = k0+{0..7, 16..23},
// lanes 16-31 hold K = k0+{8..15, 24..31}.
__device__ __forceinline__ v16bf load_a_frag(const bf16_t* A, int ld, int row0, int k0) {
    int lane = threadIdx.x & 31;
    int m  = lane & 15;
    int hi = lane >> 4;
    const bf16_t* base = A + (size_t)(row0 + m) * ld + k0 + hi * 8;
    v8bf a0 = load8(base);        // K = k0 + hi*8      .. +7
    v8bf a1 = load8(base + 16);   // K = k0 + 16 + hi*8 .. +7
    v16bf r;
#pragma unroll
    for (int i = 0; i < 8; ++i) { r[i] = a0[i]; r[i + 8] = a1[i]; }
    return r;
}

// B-fragment (32x16, KxN) where B[k][n] = M[(n0+n)*ld + k0 + k]  (columns of B
// are contiguous rows of M). Lanes 0-15: column n, K = k0+{0..7, 16..23};
// lanes 16-31: column n, K = k0+{8..15, 24..31}.
__device__ __forceinline__ v16bf load_b_frag(const bf16_t* M, int ld, int n0, int k0) {
    int lane = threadIdx.x & 31;
    int n  = lane & 15;
    int hi = lane >> 4;
    const bf16_t* base = M + (size_t)(n0 + n) * ld + k0 + hi * 8;
    v8bf b0 = load8(base);
    v8bf b1 = load8(base + 16);
    v16bf r;
#pragma unroll
    for (int i = 0; i < 8; ++i) { r[i] = b0[i]; r[i + 8] = b1[i]; }
    return r;
}

__device__ __forceinline__ v8f wmma_bf16(v16bf a, v16bf b, v8f c) {
    return __builtin_amdgcn_wmma_f32_16x16x32_bf16(
        /*neg_a=*/false, a, /*neg_b=*/false, b,
        /*c_mod=*/(short)0, c, /*reuse_a=*/false, /*reuse_b=*/false);
}

// ---------------------------------------------------------------------------
// f32 -> bf16 truncation of inputs / weights into workspace
// ---------------------------------------------------------------------------
__global__ void cvt_f32_bf16(const float* __restrict__ in, bf16_t* __restrict__ out, int n) {
    int i = blockIdx.x * blockDim.x + threadIdx.x;
    if (i < n) out[i] = (bf16_t)in[i];
}

// ---------------------------------------------------------------------------
// QKV projection: Y = X W^T, X: [NTOK x DMODEL], W: [DMODEL x DMODEL] row-major
// ([out,in], so B-columns are W rows -> contiguous 128-bit loads).
// 32x64 wave tile, copy-free 2-stage pipelined k-loop.
// mode 0: write [b][h][s][d] (q,k);  mode 2: write [b][h][d][s] (v transposed)
// ---------------------------------------------------------------------------
__global__ __launch_bounds__(256)
void qkv_gemm(const bf16_t* __restrict__ X, const bf16_t* __restrict__ W,
              bf16_t* __restrict__ out, int mode) {
    int wv   = threadIdx.x >> 5;
    int lane = threadIdx.x & 31;
    int r0 = blockIdx.y * 256 + wv * 32;     // 8 waves x 32 rows
    int n0 = blockIdx.x * 64;

    v8f c[2][4] = {};

    v16bf a0 = load_a_frag(X, DMODEL, r0,      0);
    v16bf a1 = load_a_frag(X, DMODEL, r0 + 16, 0);
    v16bf b0 = load_b_frag(W, DMODEL, n0,      0);
    v16bf b1 = load_b_frag(W, DMODEL, n0 + 16, 0);
    v16bf b2 = load_b_frag(W, DMODEL, n0 + 32, 0);
    v16bf b3 = load_b_frag(W, DMODEL, n0 + 48, 0);

#pragma unroll 1
    for (int k0 = 0; k0 < DMODEL; k0 += 64) {
        int k1 = k0 + 32;                                // always < DMODEL
        int k2 = (k0 + 64 < DMODEL) ? (k0 + 64) : k0;    // clamp on last iter

        // stage 1: issue loads for k1, compute k0 (fragments loaded last stage)
        v16bf ta0 = load_a_frag(X, DMODEL, r0,      k1);
        v16bf ta1 = load_a_frag(X, DMODEL, r0 + 16, k1);
        v16bf tb0 = load_b_frag(W, DMODEL, n0,      k1);
        v16bf tb1 = load_b_frag(W, DMODEL, n0 + 16, k1);
        v16bf tb2 = load_b_frag(W, DMODEL, n0 + 32, k1);
        v16bf tb3 = load_b_frag(W, DMODEL, n0 + 48, k1);

        c[0][0] = wmma_bf16(a0, b0, c[0][0]);
        c[0][1] = wmma_bf16(a0, b1, c[0][1]);
        c[0][2] = wmma_bf16(a0, b2, c[0][2]);
        c[0][3] = wmma_bf16(a0, b3, c[0][3]);
        c[1][0] = wmma_bf16(a1, b0, c[1][0]);
        c[1][1] = wmma_bf16(a1, b1, c[1][1]);
        c[1][2] = wmma_bf16(a1, b2, c[1][2]);
        c[1][3] = wmma_bf16(a1, b3, c[1][3]);

        // stage 2: issue loads for k2 directly into loop-carried regs,
        // compute k1 from the temporaries (no rotation copies).
        a0 = load_a_frag(X, DMODEL, r0,      k2);
        a1 = load_a_frag(X, DMODEL, r0 + 16, k2);
        b0 = load_b_frag(W, DMODEL, n0,      k2);
        b1 = load_b_frag(W, DMODEL, n0 + 16, k2);
        b2 = load_b_frag(W, DMODEL, n0 + 32, k2);
        b3 = load_b_frag(W, DMODEL, n0 + 48, k2);

        c[0][0] = wmma_bf16(ta0, tb0, c[0][0]);
        c[0][1] = wmma_bf16(ta0, tb1, c[0][1]);
        c[0][2] = wmma_bf16(ta0, tb2, c[0][2]);
        c[0][3] = wmma_bf16(ta0, tb3, c[0][3]);
        c[1][0] = wmma_bf16(ta1, tb0, c[1][0]);
        c[1][1] = wmma_bf16(ta1, tb1, c[1][1]);
        c[1][2] = wmma_bf16(ta1, tb2, c[1][2]);
        c[1][3] = wmma_bf16(ta1, tb3, c[1][3]);
    }

    int nn = lane & 15, hi = lane >> 4;
#pragma unroll
    for (int mt = 0; mt < 2; ++mt) {
#pragma unroll
        for (int j = 0; j < 4; ++j) {
#pragma unroll
            for (int i = 0; i < 8; ++i) {
                int r   = r0 + 16 * mt + i + 8 * hi;   // token row = s*BATCH + b
                int col = n0 + 16 * j + nn;            // feature   = h*HDIM + d
                int s = r >> 2;
                int b = r & 3;
                int h = col >> 6;
                int d = col & 63;
                bf16_t val = (bf16_t)c[mt][j][i];
                if (mode == 2)
                    out[((size_t)(b * NHEADS + h) * HDIM + d) * S_LEN + s] = val;
                else
                    out[((size_t)(b * NHEADS + h) * S_LEN + s) * HDIM + d] = val;
            }
        }
    }
}

// ---------------------------------------------------------------------------
// Flash attention: one wave owns rows [s0, s0+16) of one (b,h).
// Q,K: [b][h][s][d], VT: [b][h][d][s]. Output written bf16 as [s][b][h*64+d].
// V fragments are loaded at the top of the t-loop so the softmax VALU work
// (exp + shuffle reductions) covers their latency before the PV WMMAs.
// ---------------------------------------------------------------------------
__global__ __launch_bounds__(256)
void flash_attn(const bf16_t* __restrict__ Q, const bf16_t* __restrict__ K,
                const bf16_t* __restrict__ VT, bf16_t* __restrict__ O) {
    __shared__ float lds[8 * 16 * 32];          // per-wave 16x32 P staging tile
    int wv   = threadIdx.x >> 5;
    int lane = threadIdx.x & 31;
    float* pbuf = lds + wv * (16 * 32);

    int wid = blockIdx.x * 8 + wv;
    int sb  = wid & 127;                        // S/16 = 128 row-bands
    int bh  = wid >> 7;                         // b*NHEADS + h
    int s0  = sb * 16;

    const bf16_t* qh = Q  + (size_t)bh * S_LEN * HDIM;
    const bf16_t* kh = K  + (size_t)bh * S_LEN * HDIM;
    const bf16_t* vh = VT + (size_t)bh * HDIM  * S_LEN;

    v16bf qa0 = load_a_frag(qh, HDIM, s0, 0);
    v16bf qa1 = load_a_frag(qh, HDIM, s0, 32);

    v8f o0 = {}, o1 = {}, o2 = {}, o3 = {};
    float mrow[8], lrow[8];
#pragma unroll
    for (int i = 0; i < 8; ++i) { mrow[i] = -3.0e38f; lrow[i] = 0.0f; }

    int nn = lane & 15, hi = lane >> 4;
    const float scale = 0.125f;                 // 1/sqrt(HDIM)

    for (int t0 = 0; t0 <= s0 + 15; t0 += 32) {
        // V fragments first: independent of scores, latency hidden by softmax.
        v16bf vb0 = load_b_frag(vh, S_LEN,  0, t0);
        v16bf vb1 = load_b_frag(vh, S_LEN, 16, t0);
        v16bf vb2 = load_b_frag(vh, S_LEN, 32, t0);
        v16bf vb3 = load_b_frag(vh, S_LEN, 48, t0);
        // speculative prefetch of next K tile (dropped silently if OOB)
        __builtin_prefetch(kh + (size_t)(t0 + 32 + nn) * HDIM, 0, 1);

        // ----- scores: two 16x16 tiles covering columns t0 .. t0+31 -----
        v8f sc[2];
#pragma unroll
        for (int tt = 0; tt < 2; ++tt) {
            v8f acc = {};
            v16bf kb0 = load_b_frag(kh, HDIM, t0 + 16 * tt, 0);
            v16bf kb1 = load_b_frag(kh, HDIM, t0 + 16 * tt, 32);
            acc = wmma_bf16(qa0, kb0, acc);
            acc = wmma_bf16(qa1, kb1, acc);
#pragma unroll
            for (int i = 0; i < 8; ++i) {
                int row = s0 + i + 8 * hi;
                int col = t0 + 16 * tt + nn;
                float v = acc[i] * scale;
                acc[i] = (col > row) ? -3.0e38f : v;   // causal mask
            }
            sc[tt] = acc;
        }

        // ----- online softmax: row statistics via 16-lane xor reductions -----
        float mnew[8], corr[8];
#pragma unroll
        for (int i = 0; i < 8; ++i) {
            float v = fmaxf(sc[0][i], sc[1][i]);
#pragma unroll
            for (int msk = 1; msk < 16; msk <<= 1)
                v = fmaxf(v, __shfl_xor(v, msk, 32));
            mnew[i] = fmaxf(mrow[i], v);
            corr[i] = __expf(mrow[i] - mnew[i]);
        }
#pragma unroll
        for (int tt = 0; tt < 2; ++tt)
#pragma unroll
            for (int i = 0; i < 8; ++i)
                sc[tt][i] = __expf(sc[tt][i] - mnew[i]);
#pragma unroll
        for (int i = 0; i < 8; ++i) {
            float rs = sc[0][i] + sc[1][i];
#pragma unroll
            for (int msk = 1; msk < 16; msk <<= 1)
                rs += __shfl_xor(rs, msk, 32);
            lrow[i] = lrow[i] * corr[i] + rs;
            mrow[i] = mnew[i];
        }
#pragma unroll
        for (int i = 0; i < 8; ++i) {
            o0[i] *= corr[i]; o1[i] *= corr[i];
            o2[i] *= corr[i]; o3[i] *= corr[i];
        }

        // ----- re-shape P (C-layout) -> A-fragment via per-wave LDS tile -----
#pragma unroll
        for (int tt = 0; tt < 2; ++tt)
#pragma unroll
            for (int i = 0; i < 8; ++i)
                pbuf[(i + 8 * hi) * 32 + 16 * tt + nn] = sc[tt][i];
        // same-wave DS ops are in-order: loads below see the stores above.
        v16bf pa;
        {
            const float* rp = pbuf + nn * 32 + hi * 8;   // row = lane&15
            float4 c0 = *reinterpret_cast<const float4*>(rp);
            float4 c1 = *reinterpret_cast<const float4*>(rp + 4);
            float4 c2 = *reinterpret_cast<const float4*>(rp + 16);
            float4 c3 = *reinterpret_cast<const float4*>(rp + 20);
            pa[0]=(bf16_t)c0.x;  pa[1]=(bf16_t)c0.y;  pa[2]=(bf16_t)c0.z;  pa[3]=(bf16_t)c0.w;
            pa[4]=(bf16_t)c1.x;  pa[5]=(bf16_t)c1.y;  pa[6]=(bf16_t)c1.z;  pa[7]=(bf16_t)c1.w;
            pa[8]=(bf16_t)c2.x;  pa[9]=(bf16_t)c2.y;  pa[10]=(bf16_t)c2.z; pa[11]=(bf16_t)c2.w;
            pa[12]=(bf16_t)c3.x; pa[13]=(bf16_t)c3.y; pa[14]=(bf16_t)c3.z; pa[15]=(bf16_t)c3.w;
        }

        // ----- PV: O += P @ V  (V fragments already resident) -----
        o0 = wmma_bf16(pa, vb0, o0);
        o1 = wmma_bf16(pa, vb1, o1);
        o2 = wmma_bf16(pa, vb2, o2);
        o3 = wmma_bf16(pa, vb3, o3);
    }

    // ----- normalize and scatter bf16 into [s][b][h*64+d] -----
    int h = bh % NHEADS;
    int b = bh / NHEADS;
#pragma unroll
    for (int i = 0; i < 8; ++i) {
        float inv = 1.0f / lrow[i];
        int s = s0 + i + 8 * hi;
        size_t base = ((size_t)s * BATCH + b) * DMODEL + h * HDIM + nn;
        O[base +  0] = (bf16_t)(o0[i] * inv);
        O[base + 16] = (bf16_t)(o1[i] * inv);
        O[base + 32] = (bf16_t)(o2[i] * inv);
        O[base + 48] = (bf16_t)(o3[i] * inv);
    }
}

// ---------------------------------------------------------------------------
// Output projection: d_out = attn @ Wo^T, fp32 result. Same 32x64 wave tile
// copy-free pipelined k-loop as qkv_gemm.
// ---------------------------------------------------------------------------
__global__ __launch_bounds__(256)
void out_proj(const bf16_t* __restrict__ A, const bf16_t* __restrict__ W,
              float* __restrict__ Y) {
    int wv   = threadIdx.x >> 5;
    int lane = threadIdx.x & 31;
    int r0 = blockIdx.y * 256 + wv * 32;
    int n0 = blockIdx.x * 64;

    v8f c[2][4] = {};

    v16bf a0 = load_a_frag(A, DMODEL, r0,      0);
    v16bf a1 = load_a_frag(A, DMODEL, r0 + 16, 0);
    v16bf b0 = load_b_frag(W, DMODEL, n0,      0);
    v16bf b1 = load_b_frag(W, DMODEL, n0 + 16, 0);
    v16bf b2 = load_b_frag(W, DMODEL, n0 + 32, 0);
    v16bf b3 = load_b_frag(W, DMODEL, n0 + 48, 0);

#pragma unroll 1
    for (int k0 = 0; k0 < DMODEL; k0 += 64) {
        int k1 = k0 + 32;
        int k2 = (k0 + 64 < DMODEL) ? (k0 + 64) : k0;

        v16bf ta0 = load_a_frag(A, DMODEL, r0,      k1);
        v16bf ta1 = load_a_frag(A, DMODEL, r0 + 16, k1);
        v16bf tb0 = load_b_frag(W, DMODEL, n0,      k1);
        v16bf tb1 = load_b_frag(W, DMODEL, n0 + 16, k1);
        v16bf tb2 = load_b_frag(W, DMODEL, n0 + 32, k1);
        v16bf tb3 = load_b_frag(W, DMODEL, n0 + 48, k1);

        c[0][0] = wmma_bf16(a0, b0, c[0][0]);
        c[0][1] = wmma_bf16(a0, b1, c[0][1]);
        c[0][2] = wmma_bf16(a0, b2, c[0][2]);
        c[0][3] = wmma_bf16(a0, b3, c[0][3]);
        c[1][0] = wmma_bf16(a1, b0, c[1][0]);
        c[1][1] = wmma_bf16(a1, b1, c[1][1]);
        c[1][2] = wmma_bf16(a1, b2, c[1][2]);
        c[1][3] = wmma_bf16(a1, b3, c[1][3]);

        a0 = load_a_frag(A, DMODEL, r0,      k2);
        a1 = load_a_frag(A, DMODEL, r0 + 16, k2);
        b0 = load_b_frag(W, DMODEL, n0,      k2);
        b1 = load_b_frag(W, DMODEL, n0 + 16, k2);
        b2 = load_b_frag(W, DMODEL, n0 + 32, k2);
        b3 = load_b_frag(W, DMODEL, n0 + 48, k2);

        c[0][0] = wmma_bf16(ta0, tb0, c[0][0]);
        c[0][1] = wmma_bf16(ta0, tb1, c[0][1]);
        c[0][2] = wmma_bf16(ta0, tb2, c[0][2]);
        c[0][3] = wmma_bf16(ta0, tb3, c[0][3]);
        c[1][0] = wmma_bf16(ta1, tb0, c[1][0]);
        c[1][1] = wmma_bf16(ta1, tb1, c[1][1]);
        c[1][2] = wmma_bf16(ta1, tb2, c[1][2]);
        c[1][3] = wmma_bf16(ta1, tb3, c[1][3]);
    }

    int nn = lane & 15, hi = lane >> 4;
#pragma unroll
    for (int mt = 0; mt < 2; ++mt)
#pragma unroll
        for (int j = 0; j < 4; ++j)
#pragma unroll
            for (int i = 0; i < 8; ++i)
                Y[(size_t)(r0 + 16 * mt + i + 8 * hi) * DMODEL + n0 + 16 * j + nn]
                    = c[mt][j][i];
}

// ---------------------------------------------------------------------------
extern "C" void kernel_launch(void* const* d_in, const int* in_sizes, int n_in,
                              void* d_out, int out_size, void* d_ws, size_t ws_size,
                              hipStream_t stream) {
    const float* x  = (const float*)d_in[0];
    const float* Wq = (const float*)d_in[1];
    const float* Wk = (const float*)d_in[2];
    const float* Wv = (const float*)d_in[3];
    const float* Wo = (const float*)d_in[4];
    float* out = (float*)d_out;

    const size_t nTokD = (size_t)NTOK * DMODEL;        // 6,291,456 elements
    const size_t nW    = (size_t)DMODEL * DMODEL;      //   589,824 elements

    char* ws = (char*)d_ws;
    size_t off = 0;
    bf16_t* xb   = (bf16_t*)(ws + off); off += nTokD * sizeof(bf16_t);
    bf16_t* wqb  = (bf16_t*)(ws + off); off += nW * sizeof(bf16_t);
    bf16_t* wkb  = (bf16_t*)(ws + off); off += nW * sizeof(bf16_t);
    bf16_t* wvb  = (bf16_t*)(ws + off); off += nW * sizeof(bf16_t);
    bf16_t* wob  = (bf16_t*)(ws + off); off += nW * sizeof(bf16_t);
    bf16_t* qbuf = (bf16_t*)(ws + off); off += nTokD * sizeof(bf16_t);  // [b][h][s][d]
    bf16_t* kbuf = (bf16_t*)(ws + off); off += nTokD * sizeof(bf16_t);  // [b][h][s][d]
    bf16_t* vbuf = (bf16_t*)(ws + off); off += nTokD * sizeof(bf16_t);  // [b][h][d][s]
    bf16_t* attn = (bf16_t*)(ws + off); off += nTokD * sizeof(bf16_t);  // [s][b][dm]
    // total workspace: ~68 MB

    // 1) bf16 conversions
    cvt_f32_bf16<<<(int)((nTokD + 255) / 256), 256, 0, stream>>>(x, xb, (int)nTokD);
    cvt_f32_bf16<<<(int)((nW + 255) / 256), 256, 0, stream>>>(Wq, wqb, (int)nW);
    cvt_f32_bf16<<<(int)((nW + 255) / 256), 256, 0, stream>>>(Wk, wkb, (int)nW);
    cvt_f32_bf16<<<(int)((nW + 255) / 256), 256, 0, stream>>>(Wv, wvb, (int)nW);
    cvt_f32_bf16<<<(int)((nW + 255) / 256), 256, 0, stream>>>(Wo, wob, (int)nW);

    // 2) Q/K/V projections (WMMA, 32x64 wave tiles -> grid (12, 32))
    dim3 gproj(DMODEL / 64, NTOK / 256);
    qkv_gemm<<<gproj, 256, 0, stream>>>(xb, wqb, qbuf, 0);
    qkv_gemm<<<gproj, 256, 0, stream>>>(xb, wkb, kbuf, 0);
    qkv_gemm<<<gproj, 256, 0, stream>>>(xb, wvb, vbuf, 2);

    // 3) causal flash attention: 4*12*128 = 6144 waves, 8 waves/block
    flash_attn<<<(BATCH * NHEADS * (S_LEN / 16)) / 8, 256, 0, stream>>>(
        qbuf, kbuf, vbuf, attn);

    // 4) output projection to fp32 d_out
    out_proj<<<gproj, 256, 0, stream>>>(attn, wob, out);
}
